// SparseOverlayLinear_10471130267906
// MI455X (gfx1250) — compile-verified, matrix operand order checked by reading, not run on hardware
//
#include <hip/hip_runtime.h>
#include <hip/hip_fp16.h>

typedef __attribute__((ext_vector_type(16))) _Float16 v16h;
typedef __attribute__((ext_vector_type(8)))  _Float16 v8h;
typedef __attribute__((ext_vector_type(8)))  float    v8f;
typedef __attribute__((ext_vector_type(4)))  float    v4f;

#define DIN   1024
#define DOUT  1024
#define MTOT  16384   // B*S = 4*4096

#define BM 128
#define BN 256
#define BK 32
#define LDS_STRIDE 40   // halfs per row (32 + 8 pad) -> 80B pitch, conflict-free b128 frag reads

// ---------------------------------------------------------------------------
// Kernel 1: wT[n][k] = fp16( float(fp16(W[k][n])) * mask24 * scale )
// 2:4 sparsity groups run along DOUT (last dim of W), exactly as the reference:
// thresh = 2nd-largest |fp16(w)| in each group of 4, keep a >= thresh (ties kept).
// Output is transposed so the GEMM's B-fragment loads are contiguous.
// ---------------------------------------------------------------------------
__global__ void sparsify24_transpose(const float* __restrict__ w,
                                     const float* __restrict__ scale_p,
                                     _Float16* __restrict__ wT) {
  int idx = blockIdx.x * blockDim.x + threadIdx.x;  // one thread per group of 4
  int k = idx >> 8;        // row of W (DIN),  256 groups per row
  int g = idx & 255;       // group index along DOUT
  const float s = scale_p[0];

  v4f wv = *(const v4f*)(w + (size_t)k * DOUT + g * 4);
  _Float16 h[4];
  float a[4];
#pragma unroll
  for (int i = 0; i < 4; ++i) {
    h[i] = (_Float16)wv[i];                 // fp16 cast first (reference order)
    a[i] = __builtin_fabsf((float)h[i]);    // fp16 values are exact in fp32
  }
  // ascending sort-network of 4; thresh = element [2] (2nd largest)
  float s0 = a[0], s1 = a[1], s2 = a[2], s3 = a[3], t;
  if (s0 > s1) { t = s0; s0 = s1; s1 = t; }
  if (s2 > s3) { t = s2; s2 = s3; s3 = t; }
  if (s0 > s2) { t = s0; s0 = s2; s2 = t; }
  if (s1 > s3) { t = s1; s1 = s3; s3 = t; }
  if (s1 > s2) { t = s1; s1 = s2; s2 = t; }
  float thresh = s2;

#pragma unroll
  for (int i = 0; i < 4; ++i) {
    float kept = (a[i] >= thresh) ? ((float)h[i] * s) : 0.0f;  // fp32 scale, then
    wT[(size_t)(g * 4 + i) * DIN + k] = (_Float16)kept;        // fp16 for the MM
  }
}

// ---------------------------------------------------------------------------
// Kernel 2: C[M,N] = fp16(x)[M,K] * wT^T[K,N] + bias, fp32 accumulate.
// Block tile 128x256, 256 threads = 8 waves (wave32) in a 2(M) x 4(N) grid.
// Each wave owns a 64x64 tile = 4x4 fragments of v_wmma_f32_16x16x32_f16:
// 16 ds_load_b128 feed 16 WMMAs -> 32 FLOP per LDS byte (2x headroom over
// the 32x64 shape). Ping-pong double-buffered LDS: one barrier per K-tile,
// next tile's global loads issue before this tile's WMMAs.
// ---------------------------------------------------------------------------
__global__ __launch_bounds__(256)
void wmma_gemm_f16(const float* __restrict__ x,
                   const _Float16* __restrict__ wT,
                   const float* __restrict__ bias,
                   float* __restrict__ out) {
  __shared__ _Float16 As[2][BM * LDS_STRIDE];   // 2 x 10 KB
  __shared__ _Float16 Bs[2][BN * LDS_STRIDE];   // 2 x 20 KB  (B stored as [n][k])

  const int tid  = threadIdx.x;
  const int wid  = tid >> 5;
  const int lane = tid & 31;
  const int hi   = lane >> 4;      // half-wave -> K phase of the fragment
  const int r16  = lane & 15;      // row/col within 16x16 tile
  const int wave_m = wid & 1;      // 2 waves along M (64 rows each)
  const int wave_n = wid >> 1;     // 4 waves along N (64 cols each)

  const int bm = blockIdx.y;
  const int bn = blockIdx.x;

  const float*    Ag = x  + (size_t)bm * BM * DIN;
  const _Float16* Bg = wT + (size_t)bn * BN * DIN;

  // cooperative tile-load mapping: 2 threads per row-segment, 16 elems each.
  // A tile: 128 rows (one segment per thread); B tile: 256 rows (two per thread).
  const int lrow = tid >> 1;          // 0..127
  const int lcol = (tid & 1) * 16;    // 0 or 16 (element offset within BK)

  v8f acc[4][4] = {};                 // 128 VGPRs of fp32 accumulators

  // ---- prologue: fetch tile 0 into buffer 0 ----
  {
    const float* ap = Ag + (size_t)lrow * DIN + lcol;
    v4f a0 = *(const v4f*)(ap);
    v4f a1 = *(const v4f*)(ap + 4);
    v4f a2 = *(const v4f*)(ap + 8);
    v4f a3 = *(const v4f*)(ap + 12);
    v8h ha, hb;
#pragma unroll
    for (int i = 0; i < 4; ++i) {
      ha[i]     = (_Float16)a0[i];
      ha[i + 4] = (_Float16)a1[i];
      hb[i]     = (_Float16)a2[i];
      hb[i + 4] = (_Float16)a3[i];
    }
    const _Float16* bp0 = Bg + (size_t)lrow * DIN + lcol;
    const _Float16* bp1 = Bg + (size_t)(lrow + 128) * DIN + lcol;
    v8h b0 = *(const v8h*)(bp0);
    v8h b1 = *(const v8h*)(bp0 + 8);
    v8h b2 = *(const v8h*)(bp1);
    v8h b3 = *(const v8h*)(bp1 + 8);
    *(v8h*)(&As[0][lrow * LDS_STRIDE + lcol])             = ha;
    *(v8h*)(&As[0][lrow * LDS_STRIDE + lcol + 8])         = hb;
    *(v8h*)(&Bs[0][lrow * LDS_STRIDE + lcol])             = b0;
    *(v8h*)(&Bs[0][lrow * LDS_STRIDE + lcol + 8])         = b1;
    *(v8h*)(&Bs[0][(lrow + 128) * LDS_STRIDE + lcol])     = b2;
    *(v8h*)(&Bs[0][(lrow + 128) * LDS_STRIDE + lcol + 8]) = b3;
  }
  __syncthreads();

  int p = 0;
  for (int kt = 0; kt < DIN; kt += BK, p ^= 1) {
    const bool has_next = (kt + BK) < DIN;

    // ---- issue global loads for tile kt+BK (latency hidden by WMMAs) ----
    v8h ha, hb, b0, b1, b2, b3;
    if (has_next) {
      const float* ap = Ag + (size_t)lrow * DIN + kt + BK + lcol;
      v4f a0 = *(const v4f*)(ap);
      v4f a1 = *(const v4f*)(ap + 4);
      v4f a2 = *(const v4f*)(ap + 8);
      v4f a3 = *(const v4f*)(ap + 12);
#pragma unroll
      for (int i = 0; i < 4; ++i) {
        ha[i]     = (_Float16)a0[i];
        ha[i + 4] = (_Float16)a1[i];
        hb[i]     = (_Float16)a2[i];
        hb[i + 4] = (_Float16)a3[i];
      }
      const _Float16* bp0 = Bg + (size_t)lrow * DIN + kt + BK + lcol;
      const _Float16* bp1 = Bg + (size_t)(lrow + 128) * DIN + kt + BK + lcol;
      b0 = *(const v8h*)(bp0);
      b1 = *(const v8h*)(bp0 + 8);
      b2 = *(const v8h*)(bp1);
      b3 = *(const v8h*)(bp1 + 8);
    }

    // ---- build fragments from LDS buffer p (two aligned b128 reads each) ----
    // frag element h: K = (h&8)*2 + (h&7) + 8*hi -> runs [8*hi,+8) and [16+8*hi,+8)
    const _Float16* Ab = As[p];
    const _Float16* Bb = Bs[p];
    v16h aF[4], bF[4];
#pragma unroll
    for (int sx = 0; sx < 4; ++sx) {
      int row = wave_m * 64 + sx * 16 + r16;
      v8h lo = *(const v8h*)(&Ab[row * LDS_STRIDE + 8 * hi]);
      v8h up = *(const v8h*)(&Ab[row * LDS_STRIDE + 16 + 8 * hi]);
      v16h f = __builtin_shufflevector(lo, up, 0,1,2,3,4,5,6,7,8,9,10,11,12,13,14,15);
      aF[sx] = f;
    }
#pragma unroll
    for (int tx = 0; tx < 4; ++tx) {
      int row = wave_n * 64 + tx * 16 + r16;
      v8h lo = *(const v8h*)(&Bb[row * LDS_STRIDE + 8 * hi]);
      v8h up = *(const v8h*)(&Bb[row * LDS_STRIDE + 16 + 8 * hi]);
      v16h f = __builtin_shufflevector(lo, up, 0,1,2,3,4,5,6,7,8,9,10,11,12,13,14,15);
      bF[tx] = f;
    }

    // ---- 16 WMMAs: D = A x B + C, fp32 accumulate ----
#pragma unroll
    for (int sx = 0; sx < 4; ++sx)
#pragma unroll
      for (int tx = 0; tx < 4; ++tx)
        acc[sx][tx] = __builtin_amdgcn_wmma_f32_16x16x32_f16(
            false, aF[sx], false, bF[tx], (short)0, acc[sx][tx], false, false);

    // ---- stage tile kt+BK into the other buffer, single barrier per iter ----
    if (has_next) {
      _Float16* An = As[p ^ 1];
      _Float16* Bn = Bs[p ^ 1];
      *(v8h*)(&An[lrow * LDS_STRIDE + lcol])             = ha;
      *(v8h*)(&An[lrow * LDS_STRIDE + lcol + 8])         = hb;
      *(v8h*)(&Bn[lrow * LDS_STRIDE + lcol])             = b0;
      *(v8h*)(&Bn[lrow * LDS_STRIDE + lcol + 8])         = b1;
      *(v8h*)(&Bn[(lrow + 128) * LDS_STRIDE + lcol])     = b2;
      *(v8h*)(&Bn[(lrow + 128) * LDS_STRIDE + lcol + 8]) = b3;
    }
    __syncthreads();
  }

  // ---- epilogue: C element j -> row j + 8*hi, col r16 (per ISA C/D layout) ----
#pragma unroll
  for (int tx = 0; tx < 4; ++tx) {
    int col = bn * BN + wave_n * 64 + tx * 16 + r16;
    float bv = bias[col];
#pragma unroll
    for (int sx = 0; sx < 4; ++sx) {
      int row0 = bm * BM + wave_m * 64 + sx * 16 + 8 * hi;
#pragma unroll
      for (int j = 0; j < 8; ++j)
        out[(size_t)(row0 + j) * DOUT + col] = acc[sx][tx][j] + bv;
    }
  }
}

extern "C" void kernel_launch(void* const* d_in, const int* in_sizes, int n_in,
                              void* d_out, int out_size, void* d_ws, size_t ws_size,
                              hipStream_t stream) {
  const float* x     = (const float*)d_in[0];   // [4,4096,1024] fp32
  const float* w     = (const float*)d_in[1];   // [1024,1024]   fp32
  const float* bias  = (const float*)d_in[2];   // [1024]        fp32
  const float* scale = (const float*)d_in[3];   // [1]           fp32
  float* out = (float*)d_out;                   // [4,4096,1024] fp32

  _Float16* wT = (_Float16*)d_ws;               // 2 MB: transposed sparse fp16 weight

  // 1) sparsify(2:4 along DOUT) + scale + cast + transpose: 1024*256 groups
  sparsify24_transpose<<<dim3((DIN * (DOUT / 4)) / 256), dim3(256), 0, stream>>>(
      w, scale, wT);

  // 2) WMMA GEMM: grid x = N-blocks (4), y = M-blocks (128); only 4 passes of
  //    x re-reads, all hitting the 192 MB L2.
  wmma_gemm_f16<<<dim3(DOUT / BN, MTOT / BM), dim3(256), 0, stream>>>(
      x, wT, bias, out);
}
